// GraphTransformer_22179211117436
// MI455X (gfx1250) — matrix operand order repaired and checked by reading, not compiled, not run on hardware
//
#include <hip/hip_runtime.h>

// ---------------------------------------------------------------------------
// Types for CDNA5 WMMA (wave32, 16x16x32 bf16 -> f32 accumulate)
// ---------------------------------------------------------------------------
typedef __attribute__((ext_vector_type(16))) __bf16 v16bf;
typedef __attribute__((ext_vector_type(8)))  __bf16 v8bf;
typedef __attribute__((ext_vector_type(4)))  __bf16 bf4;
typedef __attribute__((ext_vector_type(8)))  float  v8f;
typedef __attribute__((ext_vector_type(4)))  float  f4;

#define TB_M 64
#define TB_N 64
#define TB_K 32
#define LDP  40   // padded LDS row pitch in bf16 elements (80B = 16B aligned)

// ---------------------------------------------------------------------------
// Generic GEMM: C = epilogue(A[M,K] @ W[K,N] + bias [+ resid])
// Requires K % 4 == 0 and N % 4 == 0 (true for every GEMM in this model).
// Software-pipelined: double-buffered LDS, global loads for tile k+1 issued
// into registers before the WMMAs of tile k, one barrier per K-step.
// mask_mode: 0 = none, 1 = node rows (mask[row]), 2 = edge rows
//            (row = b*16384 + i*128 + j -> mask[b*128+i]*mask[b*128+j])
// ---------------------------------------------------------------------------
__global__ __launch_bounds__(128) void gemm_bf16_kernel(
    const float* __restrict__ A, const float* __restrict__ W,
    const float* __restrict__ bias, const float* __restrict__ resid,
    const float* __restrict__ nmask, float* __restrict__ C,
    int M, int N, int K, int act, int mask_mode)
{
  __shared__ __align__(16) __bf16 As[2][TB_M * LDP];
  __shared__ __align__(16) __bf16 Bs[2][TB_N * LDP];
  const int t    = threadIdx.x;
  const int lane = t & 31;
  const int wave = t >> 5;
  const int half = lane >> 4;   // 0: lanes 0-15, 1: lanes 16-31
  const int lrow = lane & 15;
  const int wm   = wave >> 1;
  const int wn   = wave & 1;
  const int row0 = blockIdx.y * TB_M;
  const int col0 = blockIdx.x * TB_N;

  v8f acc[2][2] = {};
  // Staging registers: all 8 b128 loads have independent destinations so they
  // pipeline with a single wait when consumed.
  f4 va[4], vb[4];
  bool oka[4], okb[4];

  // --- phase 1: issue global loads for one K-tile into registers ---
  auto stage_load = [&](int k0) {
#pragma unroll
    for (int i = 0; i < 4; ++i) {          // A tile: 64x32, 4 float4/thread
      int idx = t + i * 128;               // 0..511
      int m  = idx >> 3;
      int k4 = (idx & 7) << 2;
      int gr = row0 + m, gk = k0 + k4;
      oka[i] = (gr < M) && (gk < K);
      size_t o = oka[i] ? ((size_t)gr * K + gk) : 0;
      va[i] = *(const f4*)(A + o);
    }
#pragma unroll
    for (int i = 0; i < 4; ++i) {          // W tile: 32x64, 4 float4/thread
      int idx = t + i * 128;
      int k  = idx >> 4;
      int n4 = (idx & 15) << 2;
      int gk = k0 + k, gn = col0 + n4;
      okb[i] = (gk < K) && (gn < N);
      size_t o = okb[i] ? ((size_t)gk * N + gn) : 0;
      vb[i] = *(const f4*)(W + o);
    }
  };

  // --- phase 2: convert fp32 -> bf16 and commit registers to LDS buffer ---
  auto stage_store = [&](int buf) {
#pragma unroll
    for (int i = 0; i < 4; ++i) {
      int idx = t + i * 128;
      int m  = idx >> 3;
      int k4 = (idx & 7) << 2;
      bf4 h;
      h.x = (__bf16)(oka[i] ? va[i].x : 0.0f);
      h.y = (__bf16)(oka[i] ? va[i].y : 0.0f);
      h.z = (__bf16)(oka[i] ? va[i].z : 0.0f);
      h.w = (__bf16)(oka[i] ? va[i].w : 0.0f);
      *(bf4*)(&As[buf][m * LDP + k4]) = h;  // 8B aligned -> ds_store_b64
    }
#pragma unroll
    for (int i = 0; i < 4; ++i) {
      int idx = t + i * 128;
      int k  = idx >> 4;
      int n4 = (idx & 15) << 2;
      // transposed store Bs[n][k] so fragment reads are contiguous K-runs
      Bs[buf][(n4 + 0) * LDP + k] = (__bf16)(okb[i] ? vb[i].x : 0.0f);
      Bs[buf][(n4 + 1) * LDP + k] = (__bf16)(okb[i] ? vb[i].y : 0.0f);
      Bs[buf][(n4 + 2) * LDP + k] = (__bf16)(okb[i] ? vb[i].z : 0.0f);
      Bs[buf][(n4 + 3) * LDP + k] = (__bf16)(okb[i] ? vb[i].w : 0.0f);
    }
  };

  // --- compute: build fragments per ISA 7.12.2 layouts, 4 WMMAs ---
  auto compute = [&](int buf) {
    v16bf afr[2], bfr[2];
#pragma unroll
    for (int mt = 0; mt < 2; ++mt) {
      int m = wm * 32 + mt * 16 + lrow;
      // A 16x32 bf16: lane half h, row m: K = {8h..8h+7, 16+8h..16+8h+7}
      v8bf c0 = *(const v8bf*)(&As[buf][m * LDP + half * 8]);
      v8bf c1 = *(const v8bf*)(&As[buf][m * LDP + 16 + half * 8]);
      afr[mt] = __builtin_shufflevector(c0, c1, 0,1,2,3,4,5,6,7,8,9,10,11,12,13,14,15);
    }
#pragma unroll
    for (int nt = 0; nt < 2; ++nt) {
      int n = wn * 32 + nt * 16 + lrow;
      // B 32x16 bf16: lane holds column n, K = {8h..8h+7, 16+8h..16+8h+7}
      v8bf c0 = *(const v8bf*)(&Bs[buf][n * LDP + half * 8]);
      v8bf c1 = *(const v8bf*)(&Bs[buf][n * LDP + 16 + half * 8]);
      bfr[nt] = __builtin_shufflevector(c0, c1, 0,1,2,3,4,5,6,7,8,9,10,11,12,13,14,15);
    }
#pragma unroll
    for (int mt = 0; mt < 2; ++mt)
#pragma unroll
      for (int nt = 0; nt < 2; ++nt)
        acc[mt][nt] = __builtin_amdgcn_wmma_f32_16x16x32_bf16(
            false, afr[mt], false, bfr[nt], (short)0, acc[mt][nt], false, false);
  };

  // --- pipelined main loop ---
  const int nk = (K + TB_K - 1) / TB_K;   // uniform across block
  stage_load(0);
  stage_store(0);
  __syncthreads();
  for (int kt = 0; kt < nk; ++kt) {
    int  cur  = kt & 1;
    bool more = (kt + 1) < nk;
    if (more) stage_load((kt + 1) * TB_K);   // overlap with WMMA below
    compute(cur);
    if (more) {
      stage_store(1 - cur);
      __syncthreads();
    }
  }

  // Epilogue. C/D layout: lane writes col = n-tile + lrow, rows half*8+v.
#pragma unroll
  for (int mt = 0; mt < 2; ++mt) {
#pragma unroll
    for (int nt = 0; nt < 2; ++nt) {
      int col = col0 + wn * 32 + nt * 16 + lrow;
      if (col >= N) continue;
      float bv = bias ? bias[col] : 0.0f;
#pragma unroll
      for (int v = 0; v < 8; ++v) {
        int row = row0 + wm * 32 + mt * 16 + half * 8 + v;
        if (row >= M) continue;
        float x = acc[mt][nt][v] + bv;
        if (resid) x += resid[(size_t)row * N + col];
        if (act) x = fmaxf(x, 0.0f);
        if (mask_mode == 1) {
          x *= nmask[row];
        } else if (mask_mode == 2) {
          int b = row >> 14, i = (row >> 7) & 127, j = row & 127;
          x *= nmask[b * 128 + i] * nmask[b * 128 + j];
        }
        C[(size_t)row * N + col] = x;
      }
    }
  }
}

// ---------------------------------------------------------------------------
// LayerNorm (wave per row) with optional fused residual: out = LN(x + res)
// ---------------------------------------------------------------------------
__global__ void ln_kernel(const float* __restrict__ x, const float* __restrict__ res,
                          const float* __restrict__ g, const float* __restrict__ b,
                          float* __restrict__ out, int rows, int L)
{
  int row  = blockIdx.x * (blockDim.x >> 5) + (threadIdx.x >> 5);
  int lane = threadIdx.x & 31;
  if (row >= rows) return;
  const float* xr = x + (size_t)row * L;
  const float* rr = res ? res + (size_t)row * L : nullptr;
  float s = 0.f, ss = 0.f;
  for (int c = lane; c < L; c += 32) {
    float v = xr[c] + (rr ? rr[c] : 0.f);
    s += v; ss += v * v;
  }
  for (int off = 16; off; off >>= 1) {
    s  += __shfl_xor(s,  off, 32);
    ss += __shfl_xor(ss, off, 32);
  }
  float mu  = s / L;
  float var = ss / L - mu * mu;
  float inv = rsqrtf(var + 1e-5f);
  float* orow = out + (size_t)row * L;
  for (int c = lane; c < L; c += 32) {
    float v = xr[c] + (rr ? rr[c] : 0.f);
    orow[c] = (v - mu) * inv * g[c] + b[c];
  }
}

// ---------------------------------------------------------------------------
// Y = Q_i * K_j / sqrt(df) * (E1 + 1) + E2   (in place over E1 buffer)
// ---------------------------------------------------------------------------
__global__ void ewy_kernel(float* __restrict__ Y, const float* __restrict__ E2,
                           const float* __restrict__ Q, const float* __restrict__ Kx)
{
  size_t idx = (size_t)blockIdx.x * blockDim.x + threadIdx.x;
  const size_t total = (size_t)4 * 128 * 128 * 256;
  if (idx >= total) return;
  int c = (int)(idx % 256);
  size_t r  = idx / 256;        // b*128*128 + i*128 + j
  int j     = (int)(r % 128);
  size_t bi = r / 128;          // b*128 + i
  int b     = (int)(bi >> 7);
  float q  = Q[bi * 256 + c];
  float k  = Kx[((size_t)b * 128 + j) * 256 + c];
  float e1 = Y[idx];
  float e2 = E2[idx];
  Y[idx] = q * k * 0.17677669529663687f * (e1 + 1.0f) + e2;
}

// ---------------------------------------------------------------------------
// FiLM broadcast: out[r,c] = addb[b,c] + (mulb[b,c] + 1) * in[r,c]
// ---------------------------------------------------------------------------
__global__ void film_kernel(const float* __restrict__ in, const float* __restrict__ addb,
                            const float* __restrict__ mulb, float* __restrict__ out,
                            size_t rows, int C, int rows_per_batch)
{
  size_t idx = (size_t)blockIdx.x * blockDim.x + threadIdx.x;
  if (idx >= rows * (size_t)C) return;
  int c = (int)(idx % C);
  size_t r = idx / C;
  int b = (int)(r / rows_per_batch);
  out[idx] = addb[b * C + c] + (mulb[b * C + c] + 1.0f) * in[idx];
}

// ---------------------------------------------------------------------------
// out[b,i,j,c] = 0.5*(in[b,i,j,c] + in[b,j,i,c]) * xm[b,i]*xm[b,j] (* (i!=j))
// ---------------------------------------------------------------------------
__global__ void sym_kernel(const float* __restrict__ in, const float* __restrict__ nmask,
                           float* __restrict__ out, int C, int zero_diag)
{
  size_t idx = (size_t)blockIdx.x * blockDim.x + threadIdx.x;
  size_t total = (size_t)4 * 128 * 128 * C;
  if (idx >= total) return;
  int c = (int)(idx % C);
  size_t r = idx / C;
  int j = (int)(r % 128);
  size_t t2 = r / 128;
  int i = (int)(t2 % 128);
  int b = (int)(t2 / 128);
  float a = in[idx];
  float s = in[(((size_t)b * 128 + j) * 128 + i) * C + c];
  float m = nmask[b * 128 + i] * nmask[b * 128 + j];
  float v = 0.5f * (a + s) * m;
  if (zero_diag && i == j) v = 0.0f;
  out[idx] = v;
}

// ---------------------------------------------------------------------------
// Online softmax over keys j (per b,i,h,f) fused with weighted-V reduction.
// ---------------------------------------------------------------------------
__global__ void attn_kernel(const float* __restrict__ Y, const float* __restrict__ V,
                            const float* __restrict__ nmask, float* __restrict__ wV)
{
  int idx = blockIdx.x * blockDim.x + threadIdx.x;
  if (idx >= 4 * 128 * 256) return;
  int c  = idx % 256;
  int bi = idx / 256;        // b*128 + i
  int b  = bi >> 7;
  const float* yrow = Y + ((size_t)bi * 128) * 256 + c;
  const float* vcol = V + ((size_t)b * 128) * 256 + c;
  const float* mk   = nmask + b * 128;
  float m = -INFINITY, s = 0.0f, acc = 0.0f;
  for (int j = 0; j < 128; ++j) {
    if (mk[j] == 0.0f) continue;
    float v  = yrow[(size_t)j * 256];
    float nm = fmaxf(m, v);
    float sc = __expf(m - nm);
    float e  = __expf(v - nm);
    s   = s * sc + e;
    acc = acc * sc + e * vcol[(size_t)j * 256];
    m = nm;
  }
  wV[(size_t)bi * 256 + c] = acc / s;
}

// ---------------------------------------------------------------------------
// Pooling stats for _xtoy/_etoy: z[b] = [mean | min | max | std(ddof=1)]
// ---------------------------------------------------------------------------
__global__ void stats_kernel(const float* __restrict__ x, float* __restrict__ z,
                             int bs, int R, int C)
{
  int idx = blockIdx.x * blockDim.x + threadIdx.x;
  if (idx >= bs * C) return;
  int b = idx / C, c = idx % C;
  const float* p = x + (size_t)b * R * C + c;
  float v0 = p[0];
  float s = v0, ss = v0 * v0, mn = v0, mx = v0;
  for (int r = 1; r < R; ++r) {
    float v = p[(size_t)r * C];
    s += v; ss += v * v;
    mn = fminf(mn, v); mx = fmaxf(mx, v);
  }
  float mean = s / R;
  float var  = (ss - R * mean * mean) / (R - 1);
  float sd   = sqrtf(fmaxf(var, 0.0f));
  float* zb = z + (size_t)b * 4 * C;
  zb[c] = mean; zb[C + c] = mn; zb[2 * C + c] = mx; zb[3 * C + c] = sd;
}

// ---------------------------------------------------------------------------
// Host orchestration
// ---------------------------------------------------------------------------
struct Lin { const float* b; const float* w; };  // pytree order: 'b' < 'w'
struct LNP { const float* b; const float* g; };  // pytree order: 'b' < 'g'

extern "C" void kernel_launch(void* const* d_in, const int* in_sizes, int n_in,
                              void* d_out_, int out_size, void* d_ws, size_t ws_size,
                              hipStream_t stream)
{
  (void)in_sizes; (void)n_in; (void)out_size; (void)ws_size;
  const float* X  = (const float*)d_in[0];
  const float* E  = (const float*)d_in[1];
  const float* y  = (const float*)d_in[2];
  const float* nm = (const float*)d_in[3];
  float* dout = (float*)d_out_;

  // Params in jax pytree (alphabetical dict key) flatten order.
  int p = 4;
  auto nextLin = [&]() { Lin l; l.b = (const float*)d_in[p++]; l.w = (const float*)d_in[p++]; return l; };
  auto nextLN  = [&]() { LNP l; l.b = (const float*)d_in[p++]; l.g = (const float*)d_in[p++]; return l; };

  Lin inE0 = nextLin(), inE1 = nextLin();
  Lin inX0 = nextLin(), inX1 = nextLin();
  Lin iny0 = nextLin(), iny1 = nextLin();
  struct LayerP {
    Lin e_add, e_mul, e_out, e_y, k, q, v, x_out, x_y,
        y_e_add, y_e_mul, y_out1, y_out2, y_x_add, y_x_mul, y_y;
    Lin linE1, linE2, linX1, linX2, lin_y1, lin_y2;
    LNP normE1, normE2, normX1, normX2, norm_y1, norm_y2;
  } Lp[2];
  for (int i = 0; i < 2; ++i) {
    Lp[i].e_add = nextLin(); Lp[i].e_mul = nextLin(); Lp[i].e_out = nextLin(); Lp[i].e_y = nextLin();
    Lp[i].k = nextLin(); Lp[i].q = nextLin(); Lp[i].v = nextLin();
    Lp[i].x_out = nextLin(); Lp[i].x_y = nextLin();
    Lp[i].y_e_add = nextLin(); Lp[i].y_e_mul = nextLin();
    Lp[i].y_out1 = nextLin(); Lp[i].y_out2 = nextLin();
    Lp[i].y_x_add = nextLin(); Lp[i].y_x_mul = nextLin(); Lp[i].y_y = nextLin();
    Lp[i].linE1 = nextLin(); Lp[i].linE2 = nextLin();
    Lp[i].linX1 = nextLin(); Lp[i].linX2 = nextLin();
    Lp[i].lin_y1 = nextLin(); Lp[i].lin_y2 = nextLin();
    Lp[i].normE1 = nextLN(); Lp[i].normE2 = nextLN();
    Lp[i].normX1 = nextLN(); Lp[i].normX2 = nextLN();
    Lp[i].norm_y1 = nextLN(); Lp[i].norm_y2 = nextLN();
  }
  Lin outE0 = nextLin(), outE1 = nextLin();
  Lin outX0 = nextLin(), outX1 = nextLin();
  Lin outy0 = nextLin(), outy1 = nextLin();

  // Workspace carve-up (floats).
  float* W0 = (float*)d_ws;
  size_t off = 0;
  auto alloc = [&](size_t n) { float* q = W0 + off; off += n; return q; };
  float* EB1 = alloc((size_t)65536 * 256);  // E1 / Y (in place)
  float* EB2 = alloc((size_t)65536 * 256);  // E2 / newE_pre
  float* EH  = alloc((size_t)65536 * 64);   // Eh
  float* ET  = alloc((size_t)65536 * 128);  // E hidden (in/FFE/out)
  float* EN  = alloc((size_t)65536 * 64);   // newE / FFN-E out
  float* E8  = alloc((size_t)65536 * 8);    // Eo pre-symmetrize
  float* XH  = alloc((size_t)512 * 256);    // Xh
  float* XT  = alloc((size_t)512 * 2048);   // X hidden (in/FFX/out)
  float* XQ  = alloc((size_t)512 * 256);
  float* XK  = alloc((size_t)512 * 256);
  float* XV  = alloc((size_t)512 * 256);
  float* XW  = alloc((size_t)512 * 256);    // wV
  float* XP  = alloc((size_t)512 * 256);    // FiLM(wV)
  float* XN  = alloc((size_t)512 * 256);    // newX / FFN-X out
  float* YH  = alloc(4 * 64);
  float* YT  = alloc(4 * 2048);
  float* YE1 = alloc(4 * 256); float* YE2 = alloc(4 * 256);
  float* YX1 = alloc(4 * 256); float* YX2 = alloc(4 * 256);
  float* ZX  = alloc(4 * 1024); float* ZE = alloc(4 * 256);
  float* S1 = alloc(4 * 64); float* S2 = alloc(4 * 64);
  float* S3 = alloc(4 * 64); float* S4 = alloc(4 * 64); float* S5 = alloc(4 * 64);

  auto gemm = [&](const float* A, Lin l, float* C, int M, int N, int K,
                  int act, int mmode, const float* res) {
    dim3 g((N + TB_N - 1) / TB_N, (M + TB_M - 1) / TB_M);
    gemm_bf16_kernel<<<g, dim3(128), 0, stream>>>(A, l.w, l.b, res, nm, C,
                                                  M, N, K, act, mmode);
  };
  auto ln = [&](float* io, const float* res, LNP pn, int rows, int Lr) {
    int blocks = (rows + 3) / 4;
    ln_kernel<<<blocks, 128, 0, stream>>>(io, res, pn.g, pn.b, io, rows, Lr);
  };
  auto ew_blocks = [](size_t total) { return (unsigned)((total + 255) / 256); };

  const size_t ED = (size_t)65536 * 256;  // big edge tensor element count

  // ---- Input MLPs ----
  gemm(X, inX0, XT, 512, 256, 16, 1, 0, nullptr);
  gemm(XT, inX1, XH, 512, 256, 256, 1, 1, nullptr);      // *xm
  gemm(E, inE0, ET, 65536, 128, 8, 1, 0, nullptr);
  gemm(ET, inE1, EN, 65536, 64, 128, 1, 0, nullptr);
  sym_kernel<<<ew_blocks((size_t)65536 * 64), 256, 0, stream>>>(EN, nm, EH, 64, 0);
  gemm(y, iny0, YT, 4, 128, 12, 1, 0, nullptr);
  gemm(YT, iny1, YH, 4, 64, 128, 1, 0, nullptr);

  // ---- Transformer layers ----
  for (int li = 0; li < 2; ++li) {
    const LayerP& L = Lp[li];
    gemm(XH, L.q, XQ, 512, 256, 256, 0, 1, nullptr);
    gemm(XH, L.k, XK, 512, 256, 256, 0, 1, nullptr);
    gemm(XH, L.v, XV, 512, 256, 256, 0, 1, nullptr);
    gemm(EH, L.e_mul, EB1, 65536, 256, 64, 0, 2, nullptr);
    gemm(EH, L.e_add, EB2, 65536, 256, 64, 0, 2, nullptr);
    ewy_kernel<<<ew_blocks(ED), 256, 0, stream>>>(EB1, EB2, XQ, XK);
    gemm(YH, L.y_e_add, YE1, 4, 256, 64, 0, 0, nullptr);
    gemm(YH, L.y_e_mul, YE2, 4, 256, 64, 0, 0, nullptr);
    film_kernel<<<ew_blocks(ED), 256, 0, stream>>>(EB1, YE1, YE2, EB2,
                                                   (size_t)65536, 256, 16384);
    gemm(EB2, L.e_out, EN, 65536, 64, 256, 0, 2, nullptr);       // newE
    attn_kernel<<<ew_blocks((size_t)512 * 256), 256, 0, stream>>>(EB1, XV, nm, XW);
    gemm(YH, L.y_x_add, YX1, 4, 256, 64, 0, 0, nullptr);
    gemm(YH, L.y_x_mul, YX2, 4, 256, 64, 0, 0, nullptr);
    film_kernel<<<ew_blocks((size_t)512 * 256), 256, 0, stream>>>(XW, YX1, YX2, XP,
                                                                  (size_t)512, 256, 128);
    gemm(XP, L.x_out, XN, 512, 256, 256, 0, 1, nullptr);         // newX
    // new_y = y_y(yh) + x_y(stats(Xh)) + e_y(stats(Eh)); then out1/out2
    stats_kernel<<<4, 256, 0, stream>>>(XH, ZX, 4, 128, 256);
    stats_kernel<<<1, 256, 0, stream>>>(EH, ZE, 4, 16384, 64);
    gemm(YH, L.y_y, S1, 4, 64, 64, 0, 0, nullptr);
    gemm(ZX, L.x_y, S2, 4, 64, 1024, 0, 0, S1);
    gemm(ZE, L.e_y, S3, 4, 64, 256, 0, 0, S2);
    gemm(S3, L.y_out1, S4, 4, 64, 64, 1, 0, nullptr);
    gemm(S4, L.y_out2, S5, 4, 64, 64, 0, 0, nullptr);
    // residual + LN1
    ln(XH, XN, L.normX1, 512, 256);
    ln(EH, EN, L.normE1, 65536, 64);
    ln(YH, S5, L.norm_y1, 4, 64);
    // FFNs + LN2
    gemm(XH, L.linX1, XT, 512, 2048, 256, 1, 0, nullptr);
    gemm(XT, L.linX2, XN, 512, 256, 2048, 0, 0, nullptr);
    ln(XH, XN, L.normX2, 512, 256);
    gemm(EH, L.linE1, ET, 65536, 128, 64, 1, 0, nullptr);
    gemm(ET, L.linE2, EN, 65536, 64, 128, 0, 0, nullptr);
    ln(EH, EN, L.normE2, 65536, 64);
    gemm(YH, L.lin_y1, YT, 4, 2048, 64, 1, 0, nullptr);
    gemm(YT, L.lin_y2, S5, 4, 64, 2048, 0, 0, nullptr);
    ln(YH, S5, L.norm_y2, 4, 64);
  }

  // ---- Output heads ----
  gemm(XH, outX0, XT, 512, 256, 256, 1, 0, nullptr);
  gemm(XT, outX1, dout, 512, 16, 256, 0, 1, X);                  // (mlp+X0)*xm
  gemm(EH, outE0, ET, 65536, 128, 64, 1, 0, nullptr);
  gemm(ET, outE1, E8, 65536, 8, 128, 0, 0, E);                   // mlp+E0
  sym_kernel<<<ew_blocks((size_t)65536 * 8), 256, 0, stream>>>(E8, nm, dout + 8192, 8, 1);
  gemm(YH, outy0, YT, 4, 128, 64, 1, 0, nullptr);
  gemm(YT, outy1, dout + 8192 + 524288, 4, 12, 128, 0, 0, y);    // mlp+y0
}
